// SelectiveSSMBlock_32701880992209
// MI455X (gfx1250) — matrix operand order repaired and checked by reading, not compiled, not run on hardware
//
#include <hip/hip_runtime.h>
#include <hip/hip_bf16.h>
#include <math.h>

typedef __bf16 bf16;
typedef __attribute__((ext_vector_type(16))) __bf16 v16bf;
typedef __attribute__((ext_vector_type(8)))  float  v8f;

static constexpr int Bb   = 4;
static constexpr int Ll   = 4096;
static constexpr int Dd   = 1024;
static constexpr int Ss   = 16;
static constexpr int Mtok = Bb * Ll;   // 16384 tokens
static constexpr int NCAT = 64;        // [W_A|W_B|W_C|W_dt|pad]

__device__ __forceinline__ float softplus_f(float x) {
  if (x > 20.f)  return x;
  if (x < -20.f) return expf(x);
  return log1pf(expf(x));
}
__device__ __forceinline__ float silu_f(float x) { return x / (1.f + expf(-x)); }

// ---- CDNA5 async global->LDS copy (ASYNCcnt-tracked, no VGPR round-trip) ----
__device__ __forceinline__ void async_load_b128(unsigned lds_byte_off,
                                                unsigned glb_byte_off,
                                                const void* sbase) {
  asm volatile("global_load_async_to_lds_b128 %0, %1, %2"
               :
               : "v"(lds_byte_off), "v"(glb_byte_off), "s"(sbase)
               : "memory");
}

// ---------------- WMMA fragment loaders (ISA 7.12.2 16-bit layouts) ----------
__device__ __forceinline__ v16bf frag_a(const bf16* base, int ld) {
  const int lane = threadIdx.x & 31;
  const int r    = lane & 15;
  const int kh   = (lane >> 4) << 3;        // 0 or 8 elements
  const bf16* row = base + r * ld + kh;
  union { uint4 q[2]; v16bf v; } u;
  u.q[0] = *reinterpret_cast<const uint4*>(row);       // K = kh .. kh+7
  u.q[1] = *reinterpret_cast<const uint4*>(row + 16);  // K = kh+16 .. kh+23
  return u.v;
}
__device__ __forceinline__ v16bf frag_b(const bf16* base, int ld) {
  const int lane = threadIdx.x & 31;
  const int n    = lane & 15;
  const int kh   = (lane >> 4) << 3;        // 0 or 8 elements
  const bf16* row = base + n * ld + kh;
  union { uint4 q[2]; v16bf v; } u;
  u.q[0] = *reinterpret_cast<const uint4*>(row);       // K = kh .. kh+7
  u.q[1] = *reinterpret_cast<const uint4*>(row + 16);  // K = kh+16 .. kh+23
  return u.v;
}

// ---------------- tiled transpose + bf16 convert: WT[c*Dd+r] = W[r*Dd+c] ----
__global__ __launch_bounds__(256) void k_transpose(const float* __restrict__ src,
                                                   bf16* __restrict__ dst) {
  __shared__ bf16 t[32][33];
  const int bx = blockIdx.x * 32;
  const int by = blockIdx.y * 32;
  const int tx = threadIdx.x & 31;
  const int ty = threadIdx.x >> 5;
#pragma unroll
  for (int i = 0; i < 32; i += 8)
    t[ty + i][tx] = (bf16)src[(size_t)(by + ty + i) * Dd + bx + tx];
  __syncthreads();
#pragma unroll
  for (int i = 0; i < 32; i += 8)
    dst[(size_t)(bx + ty + i) * Dd + by + tx] = t[tx][ty + i];
}

// ---------------- build transposed concat weight: WcatT[64][1024] -----------
__global__ __launch_bounds__(256) void k_cat(
    const float* __restrict__ WA, const float* __restrict__ WB,
    const float* __restrict__ WC, const float* __restrict__ Wdt,
    bf16* __restrict__ WcatT) {
  const int idx = blockIdx.x * 256 + threadIdx.x;   // 64*1024
  const int c = idx >> 10;
  const int r = idx & 1023;
  float v = 0.f;
  if      (c < 16) v = WA[r * 16 + c];
  else if (c < 32) v = WB[r * 16 + (c - 16)];
  else if (c < 48) v = WC[r * 16 + (c - 32)];
  else if (c == 48) v = Wdt[r];
  WcatT[idx] = (bf16)v;
}

// ---------------- layernorm: one block per token ----------------------------
__global__ __launch_bounds__(256) void k_layernorm(
    const float* __restrict__ xin, const float* __restrict__ gamma,
    const float* __restrict__ beta, bf16* __restrict__ xout) {
  __shared__ float red[256];
  const int m = blockIdx.x;
  const float* row = xin + (size_t)m * Dd;
  float v[4];
  float s = 0.f;
#pragma unroll
  for (int i = 0; i < 4; ++i) { v[i] = row[threadIdx.x + i * 256]; s += v[i]; }
  red[threadIdx.x] = s; __syncthreads();
  for (int off = 128; off > 0; off >>= 1) {
    if (threadIdx.x < off) red[threadIdx.x] += red[threadIdx.x + off];
    __syncthreads();
  }
  const float mu = red[0] * (1.f / Dd);
  __syncthreads();
  float s2 = 0.f;
#pragma unroll
  for (int i = 0; i < 4; ++i) { const float d = v[i] - mu; s2 += d * d; }
  red[threadIdx.x] = s2; __syncthreads();
  for (int off = 128; off > 0; off >>= 1) {
    if (threadIdx.x < off) red[threadIdx.x] += red[threadIdx.x + off];
    __syncthreads();
  }
  const float rstd = rsqrtf(red[0] * (1.f / Dd) + 1e-5f);
  bf16* orow = xout + (size_t)m * Dd;
#pragma unroll
  for (int i = 0; i < 4; ++i) {
    const int d = threadIdx.x + i * 256;
    orow[d] = (bf16)(((v[i] - mu) * rstd) * gamma[d] + beta[d]);
  }
}

// ---------------- WMMA GEMM: out = epi(A[MxK] @ W[KxN]), W given as W^T -----
// 256 threads = 8 waves; block tile 128 x BLK_N; wave tile 32 x (BLK_N/2).
// Triple-buffered LDS fed by GLOBAL_LOAD_ASYNC_TO_LDS_B128 (ASYNCcnt),
// one barrier per K-step, peeled tail, incremental buffer bookkeeping.
// EPI 0: silu(acc + bias) -> bf16 ; EPI 1: raw f32 ; EPI 2: acc+bias+resid f32
template<int BLK_N, int EPI>
__global__ __launch_bounds__(256, 2) void k_gemm(
    const bf16* __restrict__ A, const bf16* __restrict__ WT,
    const float* __restrict__ bias, const float* __restrict__ resid,
    float* __restrict__ outF, bf16* __restrict__ outH,
    int M, int N, int K) {
  constexpr int BLK_M = 128;
  constexpr int BLK_K = 32;
  constexpr int LDT   = BLK_K + 8;                 // 40 bf16 (16B-aligned rows)
  constexpr int WN    = BLK_N / 2;
  constexpr int NT    = WN / 16;
  constexpr int MT    = 2;
  constexpr int RA    = BLK_M * BLK_K / 8 / 256;   // b128 copies/thread (A)
  constexpr int RB    = BLK_N * BLK_K / 8 / 256;   // b128 copies/thread (B^T)

  __shared__ alignas(16) bf16 sA[3][BLK_M * LDT];
  __shared__ alignas(16) bf16 sB[3][BLK_N * LDT];

  const int tid  = threadIdx.x;
  const int lane = tid & 31;
  const int wave = tid >> 5;
  const int wm   = wave & 3;
  const int wn   = wave >> 2;
  const int m0   = blockIdx.y * BLK_M;
  const int n0   = blockIdx.x * BLK_N;

  v8f acc[MT][NT];
#pragma unroll
  for (int i = 0; i < MT; ++i) {
#pragma unroll
    for (int j = 0; j < NT; ++j) {
#pragma unroll
      for (int r = 0; r < 8; ++r) acc[i][j][r] = 0.f;
    }
  }

  // LDS byte offsets (flat LDS address carries offset in low 32 bits)
  const unsigned uA = (unsigned)(uintptr_t)&sA[0][0];
  const unsigned uB = (unsigned)(uintptr_t)&sB[0][0];
  constexpr unsigned SZA = BLK_M * LDT * 2;
  constexpr unsigned SZB = BLK_N * LDT * 2;

  // per-thread chunk coordinates (constant across K-steps)
  const int ca0 = tid, ca1 = tid + 256;            // A chunk ids (RA == 2)
  const int cb0 = tid, cb1 = tid + 256;            // B chunk ids
  const unsigned aoff0 = (unsigned)(((ca0 >> 2) * LDT + (ca0 & 3) * 8) * 2);
  const unsigned aoff1 = (unsigned)(((ca1 >> 2) * LDT + (ca1 & 3) * 8) * 2);
  const unsigned goffA0 = (unsigned)((((ca0 >> 2) * K) + (ca0 & 3) * 8) * 2);
  const unsigned goffA1 = (unsigned)((((ca1 >> 2) * K) + (ca1 & 3) * 8) * 2);
  const unsigned boff0 = (unsigned)(((cb0 >> 2) * LDT + (cb0 & 3) * 8) * 2);
  const unsigned boff1 = (unsigned)(((cb1 >> 2) * LDT + (cb1 & 3) * 8) * 2);
  const unsigned goffB0 = (unsigned)((((cb0 >> 2) * K) + (cb0 & 3) * 8) * 2);
  const unsigned goffB1 = (unsigned)((((cb1 >> 2) * K) + (cb1 & 3) * 8) * 2);

  auto issue = [&](const bf16* baseA, const bf16* baseB, unsigned bufA,
                   unsigned bufB) {
    async_load_b128(bufA + aoff0, goffA0, baseA);
    if (RA > 1) async_load_b128(bufA + aoff1, goffA1, baseA);
    async_load_b128(bufB + boff0, goffB0, baseB);
    if (RB > 1) async_load_b128(bufB + boff1, goffB1, baseB);
  };

  auto compute = [&](int get) {
    v16bf af[MT], bv[NT];
#pragma unroll
    for (int i = 0; i < MT; ++i)
      af[i] = frag_a(&sA[get][(wm * 32 + i * 16) * LDT], LDT);
#pragma unroll
    for (int j = 0; j < NT; ++j)
      bv[j] = frag_b(&sB[get][(wn * WN + j * 16) * LDT], LDT);
#pragma unroll
    for (int i = 0; i < MT; ++i) {
#pragma unroll
      for (int j = 0; j < NT; ++j) {
        acc[i][j] = __builtin_amdgcn_wmma_f32_16x16x32_bf16(
            false, af[i], false, bv[j], (short)0, acc[i][j], false, false);
      }
    }
  };

  const int nk = K >> 5;
  const bf16* pA = A  + (size_t)m0 * K;            // advances 32 elems/K-step
  const bf16* pB = WT + (size_t)n0 * K;

  issue(pA, pB, uA, uB);                           // tile 0 -> buffer 0
  int get = 0;
  unsigned putA = uA + SZA, putB = uB + SZB;       // buffer 1 next

  for (int kt = 0; kt < nk - 1; ++kt) {
    pA += BLK_K; pB += BLK_K;
    issue(pA, pB, putA, putB);                     // tile kt+1
    // RA+RB just-issued copies may stay outstanding; all older (tile kt) done
    asm volatile("s_wait_asynccnt %0" :: "n"(RA + RB) : "memory");
    __builtin_prefetch(pA + (size_t)(tid >> 2) * K + BLK_K, 0, 0);
    __builtin_prefetch(pB + (size_t)(tid >> 2) * K + BLK_K, 0, 0);
    __syncthreads();                               // tile kt visible everywhere
    compute(get);
    get = (get == 2) ? 0 : get + 1;
    putA = (putA == uA + 2 * SZA) ? uA : putA + SZA;
    putB = (putB == uB + 2 * SZB) ? uB : putB + SZB;
  }
  asm volatile("s_wait_asynccnt 0" ::: "memory");
  __syncthreads();
  compute(get);                                    // last tile

  // C/D layout: vgpr r + 8*(lane>=16) = row, lane&15 = col
  const int rlo = lane & 15;
  const int kh  = lane >> 4;
#pragma unroll
  for (int i = 0; i < MT; ++i) {
#pragma unroll
    for (int j = 0; j < NT; ++j) {
      const int nb = n0 + wn * WN + j * 16 + rlo;
#pragma unroll
      for (int r = 0; r < 8; ++r) {
        const int mb = m0 + wm * 32 + i * 16 + r + kh * 8;
        const size_t off = (size_t)mb * N + nb;
        float v = acc[i][j][r];
        if constexpr (EPI == 0) {
          outH[off] = (bf16)silu_f(v + bias[nb]);
        } else if constexpr (EPI == 1) {
          outF[off] = v;
        } else {
          outF[off] = v + bias[nb] + resid[off];
        }
      }
    }
  }
}

// ---------------- u = row mean of x ------------------------------------------
__global__ __launch_bounds__(256) void k_rowmean(const bf16* __restrict__ x,
                                                 float* __restrict__ u) {
  __shared__ float red[256];
  const int m = blockIdx.x;
  const bf16* row = x + (size_t)m * Dd;
  float s = 0.f;
  for (int i = threadIdx.x; i < Dd; i += 256) s += (float)row[i];
  red[threadIdx.x] = s; __syncthreads();
  for (int off = 128; off > 0; off >>= 1) {
    if (threadIdx.x < off) red[threadIdx.x] += red[threadIdx.x + off];
    __syncthreads();
  }
  if (threadIdx.x == 0) u[m] = red[0] * (1.f / Dd);
}

// ---------------- gate: A_disc, Bu, C from projections -----------------------
__global__ __launch_bounds__(256) void k_gate(
    const float* __restrict__ proj, const float* __restrict__ u,
    const float* __restrict__ bA, const float* __restrict__ bBv,
    const float* __restrict__ bC, const float* __restrict__ bdt,
    float* __restrict__ Adisc, float* __restrict__ Buv, float* __restrict__ Cm) {
  const int g = blockIdx.x * 256 + threadIdx.x;  // [0, Mtok*16)
  const int m = g >> 4, s = g & 15;
  const float* p = proj + (size_t)m * NCAT;
  const float dt = softplus_f(p[48] + bdt[0]);
  const float la = -softplus_f(p[s] + bA[s]);
  const float bt = p[16 + s] + bBv[s];
  const float ct = p[32 + s] + bC[s];
  Adisc[g] = expf(la * dt);
  Buv[g]   = bt * dt * u[m];
  Cm[g]    = ct;
}

// ---------------- associative scan: h_t = a_t h_{t-1} + bu_t ----------------
__global__ __launch_bounds__(256) void k_scan(
    const float* __restrict__ Adisc, const float* __restrict__ Buv,
    const float* __restrict__ hprev, float* __restrict__ hs,
    float* __restrict__ hfinal) {
  __shared__ float sAo[256], sBo[256];
  const int b = blockIdx.x >> 4;
  const int s = blockIdx.x & 15;
  const int tid = threadIdx.x;
  const float h0 = hprev[b * Ss + s];
  constexpr int CH = Ll / 256;  // 16
  const size_t base = (size_t)b * Ll * Ss + s;
  const int t0 = tid * CH;

  float ac = 1.f, bc = 0.f;
#pragma unroll
  for (int i = 0; i < CH; ++i) {
    const size_t idx = base + (size_t)(t0 + i) * Ss;
    const float a = Adisc[idx], bu = Buv[idx];
    bc = bc * a + bu;
    ac = ac * a;
  }
  sAo[tid] = ac; sBo[tid] = bc; __syncthreads();
  float ca = ac, cb = bc;
  for (int off = 1; off < 256; off <<= 1) {
    float pa = 1.f, pb = 0.f;
    if (tid >= off) { pa = sAo[tid - off]; pb = sBo[tid - off]; }
    __syncthreads();
    const float na = pa * ca;
    const float nb = pb * ca + cb;          // earlier ∘ self
    sAo[tid] = na; sBo[tid] = nb;
    ca = na; cb = nb;
    __syncthreads();
  }
  float h = h0;
  if (tid > 0) h = sAo[tid - 1] * h0 + sBo[tid - 1];
#pragma unroll
  for (int i = 0; i < CH; ++i) {
    const size_t idx = base + (size_t)(t0 + i) * Ss;
    h = Adisc[idx] * h + Buv[idx];
    hs[idx] = h;
  }
  if (tid == 255) hfinal[b * Ss + s] = h;
}

// ---------------- y = sum_s C*h ; xy = x * y (bf16) --------------------------
__global__ __launch_bounds__(256) void k_y(
    const float* __restrict__ Cm, const float* __restrict__ hs,
    const bf16* __restrict__ x, bf16* __restrict__ xy) {
  const int m = blockIdx.x;
  float y = 0.f;
#pragma unroll
  for (int s2 = 0; s2 < 16; ++s2)
    y += Cm[(size_t)m * 16 + s2] * hs[(size_t)m * 16 + s2];
  const bf16* xr = x + (size_t)m * Dd;
  bf16* o = xy + (size_t)m * Dd;
  for (int d = threadIdx.x; d < Dd; d += 256) o[d] = (bf16)((float)xr[d] * y);
}

extern "C" void kernel_launch(void* const* d_in, const int* in_sizes, int n_in,
                              void* d_out, int out_size, void* d_ws, size_t ws_size,
                              hipStream_t stream) {
  const float* x_seq  = (const float*)d_in[0];
  const float* h_prev = (const float*)d_in[1];
  const float* gamma  = (const float*)d_in[2];
  const float* beta   = (const float*)d_in[3];
  const float* W_in   = (const float*)d_in[4];
  const float* b_in   = (const float*)d_in[5];
  const float* W_A    = (const float*)d_in[6];
  const float* b_A    = (const float*)d_in[7];
  const float* W_B    = (const float*)d_in[8];
  const float* b_B    = (const float*)d_in[9];
  const float* W_C    = (const float*)d_in[10];
  const float* b_C    = (const float*)d_in[11];
  const float* W_dt   = (const float*)d_in[12];
  const float* b_dt   = (const float*)d_in[13];
  const float* W_out  = (const float*)d_in[14];
  const float* b_out  = (const float*)d_in[15];

  float* yout = (float*)d_out;                       // (B,L,D)
  float* hfin = yout + (size_t)Mtok * Dd;            // (B,S)

  char* ws = (char*)d_ws;
  size_t off = 0;
  auto alloc = [&](size_t bytes) -> char* {
    char* p = ws + off;
    off += (bytes + 255) & ~(size_t)255;
    return p;
  };
  bf16*  xln   = (bf16*)alloc((size_t)Mtok * Dd * sizeof(bf16)); // reused as xy
  bf16*  xbf   = (bf16*)alloc((size_t)Mtok * Dd * sizeof(bf16));
  bf16*  WiT   = (bf16*)alloc((size_t)Dd * Dd * sizeof(bf16));   // W_in^T
  bf16*  WoT   = (bf16*)alloc((size_t)Dd * Dd * sizeof(bf16));   // W_out^T
  bf16*  WcatT = (bf16*)alloc((size_t)NCAT * Dd * sizeof(bf16)); // cat^T
  float* proj  = (float*)alloc((size_t)Mtok * NCAT * sizeof(float));
  float* u     = (float*)alloc((size_t)Mtok * sizeof(float));
  float* Adis  = (float*)alloc((size_t)Mtok * Ss * sizeof(float));
  float* Buv   = (float*)alloc((size_t)Mtok * Ss * sizeof(float));
  float* Cm    = (float*)alloc((size_t)Mtok * Ss * sizeof(float));
  float* hsb   = (float*)alloc((size_t)Mtok * Ss * sizeof(float));
  bf16*  xy    = xln;  // alias: xln dead after in-projection GEMM

  // 1) transpose + convert weights once per call
  k_transpose<<<dim3(Dd / 32, Dd / 32), 256, 0, stream>>>(W_in,  WiT);
  k_transpose<<<dim3(Dd / 32, Dd / 32), 256, 0, stream>>>(W_out, WoT);
  k_cat<<<NCAT * Dd / 256, 256, 0, stream>>>(W_A, W_B, W_C, W_dt, WcatT);
  // 2) layernorm -> bf16
  k_layernorm<<<Mtok, 256, 0, stream>>>(x_seq, gamma, beta, xln);
  // 3) x = silu(ln @ W_in + b_in)   [WMMA + async-LDS]
  k_gemm<128, 0><<<dim3(Dd / 128, Mtok / 128), 256, 0, stream>>>(
      xln, WiT, b_in, nullptr, nullptr, xbf, Mtok, Dd, Dd);
  // 4) u = mean(x)
  k_rowmean<<<Mtok, 256, 0, stream>>>(xbf, u);
  // 5) proj = x @ [W_A|W_B|W_C|W_dt]   [WMMA + async-LDS]
  k_gemm<64, 1><<<dim3(NCAT / 64, Mtok / 128), 256, 0, stream>>>(
      xbf, WcatT, nullptr, nullptr, proj, nullptr, Mtok, NCAT, Dd);
  // 6) gate math
  k_gate<<<Mtok * Ss / 256, 256, 0, stream>>>(
      proj, u, b_A, b_B, b_C, b_dt, Adis, Buv, Cm);
  // 7) selective scan over L (associative scan per (b,s))
  k_scan<<<Bb * Ss, 256, 0, stream>>>(Adis, Buv, h_prev, hsb, hfin);
  // 8) y = sum_s C*h ; xy = x*y
  k_y<<<Mtok, 256, 0, stream>>>(Cm, hsb, xbf, xy);
  // 9) y_seq = xy @ W_out + b_out + x_seq   [WMMA + async-LDS]
  k_gemm<128, 2><<<dim3(Dd / 128, Mtok / 128), 256, 0, stream>>>(
      xy, WoT, b_out, x_seq, yout, nullptr, Mtok, Dd, Dd);
}
